// ForgetfulCausalTopKAttention_64012192579838
// MI455X (gfx1250) — matrix operand order repaired and checked by reading, not compiled, not run on hardware
//
#include <hip/hip_runtime.h>

// ---------------------------------------------------------------------------
// MLA-style sliding-window + sink attention for MI455X (gfx1250).
// bf16 WMMA (v_wmma_f32_16x16x32_bf16, wave32) for every GEMM; weights are
// pre-transposed at conversion so all WMMA fragments load as contiguous
// 32-byte LDS reads.  Tile staging uses gfx1250 async global->LDS copies
// (global_load_async_to_lds_b128), double-buffered in the GEMM K-loop with
// s_wait_asynccnt threshold waits so HBM latency overlaps the WMMAs.
// ---------------------------------------------------------------------------

typedef __attribute__((ext_vector_type(16))) __bf16 v16bf;
typedef __attribute__((ext_vector_type(8)))  __bf16 v8bf;
typedef __attribute__((ext_vector_type(8)))  float  v8f;
typedef __attribute__((ext_vector_type(4)))  int    v4i;

#define NUM_HEADS 16
#define HEAD_DIM  128
#define WINDOW    128
#define PERSIST   16
#define MAXT      10          // max active key tiles per 16-query tile
#define SKEY      (MAXT * 16) // 160 staged keys

#if __has_builtin(__builtin_amdgcn_global_load_async_to_lds_b128) && \
    __has_builtin(__builtin_amdgcn_s_wait_asynccnt)
#define USE_ASYNC_LDS 1
#else
#define USE_ASYNC_LDS 0
#endif

// 16-byte global -> LDS copy (async on gfx1250 when available)
static __device__ __forceinline__ void cp16(void* lds_dst, const void* gsrc) {
#if USE_ASYNC_LDS
  __builtin_amdgcn_global_load_async_to_lds_b128(
      (__attribute__((address_space(1))) v4i*)(uintptr_t)gsrc,
      (__attribute__((address_space(3))) v4i*)(unsigned)(uintptr_t)lds_dst,
      0, 0);
#else
  *(uint4*)lds_dst = *(const uint4*)gsrc;
#endif
}

// wait until at most `n` async copies remain in flight (0 in fallback)
static __device__ __forceinline__ void async_wait_le4() {
#if USE_ASYNC_LDS
  __builtin_amdgcn_s_wait_asynccnt(4);
#endif
}
static __device__ __forceinline__ void async_wait_le0() {
#if USE_ASYNC_LDS
  __builtin_amdgcn_s_wait_asynccnt(0);
#endif
}

static __device__ __forceinline__ v16bf cat8(v8bf lo, v8bf hi) {
  return __builtin_shufflevector(lo, hi, 0, 1, 2, 3, 4, 5, 6, 7,
                                         8, 9, 10, 11, 12, 13, 14, 15);
}

static __device__ __forceinline__ v8f wmma_bf16(v16bf a, v16bf b, v8f c) {
  return __builtin_amdgcn_wmma_f32_16x16x32_bf16(
      false, a, false, b, (short)0, c, false, false);
}

// ---------------------------------------------------------------------------
// Elementwise f32 -> bf16
// ---------------------------------------------------------------------------
__global__ void cvt_f32_bf16_kernel(const float* __restrict__ in,
                                    __bf16* __restrict__ out, int n) {
  int i = blockIdx.x * blockDim.x + threadIdx.x;
  if (i < n) out[i] = (__bf16)in[i];
}

// ---------------------------------------------------------------------------
// Transposing f32 -> bf16: in[K][N]  ->  out[N][K]   (K,N multiples of 32)
// ---------------------------------------------------------------------------
__global__ __launch_bounds__(256) void cvt_transpose_kernel(
    const float* __restrict__ in, __bf16* __restrict__ out, int K, int N) {
  __shared__ float tile[32][33];
  const int k0 = blockIdx.y * 32, n0 = blockIdx.x * 32;
  const int tx = threadIdx.x, ty = threadIdx.y;   // block (32,8)
  for (int i = ty; i < 32; i += 8)
    tile[i][tx] = in[(size_t)(k0 + i) * N + (n0 + tx)];
  __syncthreads();
  for (int i = ty; i < 32; i += 8)
    out[(size_t)(n0 + i) * K + (k0 + tx)] = (__bf16)tile[tx][i];
}

// ---------------------------------------------------------------------------
// Tiled bf16 GEMM: C[M,N] = A[M,K] @ B[K,N] with B supplied TRANSPOSED
// (Bt[N][K] row-major).  Block tile 128x128, 256 threads = 8 wave32 waves
// (2x4), each wave a 64x32 sub-tile.  Double-buffered async LDS staging.
// M,N % 128 == 0, K % 32 == 0.
// ---------------------------------------------------------------------------
#define BM 128
#define BN 128
#define BK 32

template <typename OutT>
__global__ __launch_bounds__(256) void gemm_bf16_kernel(
    const __bf16* __restrict__ A, const __bf16* __restrict__ Bt,
    OutT* __restrict__ C, int M, int N, int K) {
  __shared__ alignas(64) __bf16 As[2][BM * BK];   // [buf][row m][k]
  __shared__ alignas(64) __bf16 Bs[2][BN * BK];   // [buf][col n][k]

  const int tid  = threadIdx.x;
  const int lane = tid & 31;
  const int wave = tid >> 5;
  const int wm   = wave >> 2;      // 0..1
  const int wn   = wave & 3;       // 0..3
  const int half = lane >> 4;
  const int l16  = lane & 15;
  const int tileM = blockIdx.y * BM;
  const int tileN = blockIdx.x * BN;

  // staging assignment: 16 bf16 (two 16B units) per thread per tile
  const int sr = tid >> 1, scol = (tid & 1) * 16;
  const int dstOff = sr * BK + scol;
  const __bf16* aSrc = A  + (size_t)(tileM + sr) * K + scol;
  const __bf16* bSrc = Bt + (size_t)(tileN + sr) * K + scol;

  v8f acc[4][2];
#pragma unroll
  for (int mt = 0; mt < 4; ++mt)
#pragma unroll
    for (int nt = 0; nt < 2; ++nt) acc[mt][nt] = (v8f){};

  const int T = K / BK;
  // prologue: stage tile 0 into buffer 0
  cp16(&As[0][dstOff], aSrc);  cp16(&As[0][dstOff + 8], aSrc + 8);
  cp16(&Bs[0][dstOff], bSrc);  cp16(&Bs[0][dstOff + 8], bSrc + 8);

  for (int k = 0; k < T; ++k) {
    const int cur = k & 1;
    if (k + 1 < T) {
      // stage next tile into the other buffer, keep it in flight
      const __bf16* a = aSrc + (size_t)(k + 1) * BK;
      const __bf16* b = bSrc + (size_t)(k + 1) * BK;
      const int nxt = cur ^ 1;
      cp16(&As[nxt][dstOff], a);  cp16(&As[nxt][dstOff + 8], a + 8);
      cp16(&Bs[nxt][dstOff], b);  cp16(&Bs[nxt][dstOff + 8], b + 8);
      async_wait_le4();            // tile k landed; tile k+1 still in flight
    } else {
      async_wait_le0();
    }
    __syncthreads();

    v16bf af[4];
#pragma unroll
    for (int mt = 0; mt < 4; ++mt) {
      const __bf16* p = &As[cur][(wm * 64 + mt * 16 + l16) * BK];
      af[mt] = cat8(*(const v8bf*)(p + (half << 4)),
                    *(const v8bf*)(p + 8 + (half << 4)));
    }
    v16bf bfr[2];
#pragma unroll
    for (int nt = 0; nt < 2; ++nt) {
      const __bf16* p = &Bs[cur][(wn * 32 + nt * 16 + l16) * BK + (half << 4)];
      bfr[nt] = *(const v16bf*)p;
    }

#pragma unroll
    for (int mt = 0; mt < 4; ++mt)
#pragma unroll
      for (int nt = 0; nt < 2; ++nt)
        acc[mt][nt] = wmma_bf16(af[mt], bfr[nt], acc[mt][nt]);

    __syncthreads();   // protects buffer cur from next iteration's restage
  }

#pragma unroll
  for (int mt = 0; mt < 4; ++mt) {
#pragma unroll
    for (int nt = 0; nt < 2; ++nt) {
      OutT* p = C + (size_t)(tileM + wm * 64 + mt * 16 + (half << 3)) * N
                  + (tileN + wn * 32 + nt * 16 + l16);
#pragma unroll
      for (int r = 0; r < 8; ++r) {
        *p = (OutT)acc[mt][nt][r];
        p += N;
      }
    }
  }
}

// ---------------------------------------------------------------------------
// Fused sliding-window + sink attention.
// Block per (16-query tile, head, batch); 128 threads = 4 waves.
// Active key tiles: {0} U {t-8..t}  (<= 10 tiles = 160 keys).
// LDS: Q tile, K strip (reused as transposed V strip), score strip, P strip.
// ---------------------------------------------------------------------------
__global__ __launch_bounds__(128) void attn_window_kernel(
    const __bf16* __restrict__ Qg, const __bf16* __restrict__ Kg,
    const __bf16* __restrict__ Vg, __bf16* __restrict__ Og,
    int S, int H) {
  const int t  = blockIdx.x;
  const int h  = blockIdx.y;
  const int b  = blockIdx.z;
  const int q0 = t * 16;
  const int D  = H * HEAD_DIM;

  const int tid  = threadIdx.x;
  const int lane = tid & 31;
  const int wave = tid >> 5;
  const int half = lane >> 4;
  const int l16  = lane & 15;
  const float scale = 0.088388347648318447f;   // 128^-0.5

  __shared__ alignas(64) __bf16 Ql[16 * HEAD_DIM];     // 4 KB
  __shared__ alignas(64) __bf16 KVl[SKEY * HEAD_DIM];  // 40 KB (K strip, then V^T)
  __shared__ alignas(64) float  sc[16][SKEY];          // 10 KB
  __shared__ alignas(64) __bf16 Pl[16 * SKEY];         // 5 KB
  __shared__ int ktl[MAXT];

  const int cnt = (t < MAXT - 1) ? (t + 1) : MAXT;
  if (tid < MAXT) {
    int v;
    if (t <= 8) v = (tid < cnt) ? tid : 0;
    else        v = (tid == 0) ? 0 : (t - 9 + tid);
    ktl[tid] = v;
  }
  __syncthreads();

  const __bf16* qbase = Qg + ((size_t)b * S + q0) * D + h * HEAD_DIM;
  const __bf16* kbase = Kg + (size_t)b * S * D + h * HEAD_DIM;
  const __bf16* vbase = Vg + (size_t)b * S * D + h * HEAD_DIM;

  // ---- stage Q tile (16x128) and K strip (160x128), 16B units, coalesced
  for (int u = tid; u < 16 * (HEAD_DIM / 8); u += 128) {
    const int row = u >> 4, seg = (u & 15) * 8;
    cp16(Ql + row * HEAD_DIM + seg, qbase + (size_t)row * D + seg);
  }
  for (int u = tid; u < SKEY * (HEAD_DIM / 8); u += 128) {
    const int key = u >> 4, seg = (u & 15) * 8;
    const int gkey = ktl[key >> 4] * 16 + (key & 15);
    cp16(KVl + key * HEAD_DIM + seg, kbase + (size_t)gkey * D + seg);
  }
  async_wait_le0();
  __syncthreads();

  // ---- Q fragments (4 K-chunks over hd=128)
  v16bf qf[4];
#pragma unroll
  for (int kc = 0; kc < 4; ++kc) {
    const __bf16* p = Ql + l16 * HEAD_DIM + kc * 32;
    qf[kc] = cat8(*(const v8bf*)(p + (half << 4)),
                  *(const v8bf*)(p + 8 + (half << 4)));
  }

  // ---- score tiles: Q @ K^T, scale + mask, write f32 strip
  for (int j = wave; j < cnt; j += 4) {
    v8f accf = (v8f){};
#pragma unroll
    for (int kc = 0; kc < 4; ++kc) {
      const __bf16* p = KVl + (j * 16 + l16) * HEAD_DIM + kc * 32 + (half << 4);
      accf = wmma_bf16(qf[kc], *(const v16bf*)p, accf);
    }
    const int kt = ktl[j];
#pragma unroll
    for (int r = 0; r < 8; ++r) {
      const int m  = (half << 3) + r;
      const int qi = q0 + m;
      const int ki = kt * 16 + l16;
      const bool ok = (ki <= qi) && (((qi - ki) < WINDOW) || (ki < PERSIST));
      sc[m][j * 16 + l16] = ok ? accf[r] * scale : -3.0e38f;
    }
  }
  __syncthreads();

  // ---- restage V strip TRANSPOSED (dim-major [128][160]) into KVl
  for (int u = tid; u < SKEY * (HEAD_DIM / 8); u += 128) {
    const int key = u >> 4, seg = (u & 15) * 8;
    const int gkey = ktl[key >> 4] * 16 + (key & 15);
    const v8bf v = *(const v8bf*)(vbase + (size_t)gkey * D + seg);
#pragma unroll
    for (int j = 0; j < 8; ++j)
      KVl[(seg + j) * SKEY + key] = v[j];
  }
  __syncthreads();

  // ---- row softmax (f32) -> bf16 probabilities, zero-padded to 160 cols
  if (tid < 16) {
    const int m = tid, ncol = cnt * 16;
    float mx = -3.0e38f;
    for (int c = 0; c < ncol; ++c) mx = fmaxf(mx, sc[m][c]);
    float sum = 0.f;
    for (int c = 0; c < ncol; ++c) {
      const float e = __expf(sc[m][c] - mx);
      sum += e;
      sc[m][c] = e;
    }
    const float inv = 1.0f / sum;
    for (int c = 0; c < ncol; ++c)     Pl[m * SKEY + c] = (__bf16)(sc[m][c] * inv);
    for (int c = ncol; c < SKEY; ++c)  Pl[m * SKEY + c] = (__bf16)0.0f;
  }
  __syncthreads();

  // ---- O = P (16x160) @ V (160x128); each wave owns two 16-wide n-tiles
  for (int nt = wave * 2; nt < wave * 2 + 2; ++nt) {
    v8f acc = (v8f){};
#pragma unroll
    for (int kc = 0; kc < 5; ++kc) {
      const __bf16* pp = Pl + l16 * SKEY + kc * 32;
      const v16bf pa = cat8(*(const v8bf*)(pp + (half << 4)),
                            *(const v8bf*)(pp + 8 + (half << 4)));
      const __bf16* vp = KVl + (nt * 16 + l16) * SKEY + kc * 32 + (half << 4);
      acc = wmma_bf16(pa, *(const v16bf*)vp, acc);
    }
    __bf16* ob = Og + ((size_t)b * S + q0 + (half << 3)) * D
                    + h * HEAD_DIM + nt * 16 + l16;
#pragma unroll
    for (int r = 0; r < 8; ++r) {
      *ob = (__bf16)acc[r];
      ob += D;
    }
  }
}

// ---------------------------------------------------------------------------
// Host-side launcher
// ---------------------------------------------------------------------------
extern "C" void kernel_launch(void* const* d_in, const int* in_sizes, int n_in,
                              void* d_out, int out_size, void* d_ws, size_t ws_size,
                              hipStream_t stream) {
  const float* x   = (const float*)d_in[0];   // [B,S,D]
  const float* Wq  = (const float*)d_in[1];   // [D,D]
  const float* Wkv = (const float*)d_in[2];   // [D,L]
  const float* Wk  = (const float*)d_in[3];   // [L,D]
  const float* Wv  = (const float*)d_in[4];   // [L,D]
  const float* Wo  = (const float*)d_in[5];   // [D,D]

  const int B = 2, S = 2048, D = 2048, L = 512, H = NUM_HEADS;
  const int M = B * S;                         // 4096

  char* ws = (char*)d_ws;
  size_t off = 0;
  auto carve = [&](size_t bytes) -> void* {
    void* p = ws + off;
    off += (bytes + 255) & ~(size_t)255;
    return p;
  };
  __bf16* xb   = (__bf16*)carve((size_t)M * D * 2);
  __bf16* Wqt  = (__bf16*)carve((size_t)D * D * 2);   // [N=D][K=D]
  __bf16* Wkvt = (__bf16*)carve((size_t)L * D * 2);   // [N=L][K=D]
  __bf16* Wkt  = (__bf16*)carve((size_t)D * L * 2);   // [N=D][K=L]
  __bf16* Wvt  = (__bf16*)carve((size_t)D * L * 2);   // [N=D][K=L]
  __bf16* Wot  = (__bf16*)carve((size_t)D * D * 2);   // [N=D][K=D]
  __bf16* qb   = (__bf16*)carve((size_t)M * D * 2);
  __bf16* kvb  = (__bf16*)carve((size_t)M * L * 2);
  __bf16* kb   = (__bf16*)carve((size_t)M * D * 2);
  __bf16* vb   = (__bf16*)carve((size_t)M * D * 2);
  __bf16* aob  = (__bf16*)carve((size_t)M * D * 2);

  // x: plain convert; weights: convert + transpose (always the B operand)
  cvt_f32_bf16_kernel<<<(M * D + 255) / 256, 256, 0, stream>>>(x, xb, M * D);
  auto cvtT = [&](const float* src, __bf16* dst, int K, int N) {
    dim3 grid(N / 32, K / 32), blk(32, 8);
    cvt_transpose_kernel<<<grid, blk, 0, stream>>>(src, dst, K, N);
  };
  cvtT(Wq,  Wqt,  D, D);
  cvtT(Wkv, Wkvt, D, L);
  cvtT(Wk,  Wkt,  L, D);
  cvtT(Wv,  Wvt,  L, D);
  cvtT(Wo,  Wot,  D, D);

  auto gemm_bf = [&](const __bf16* A, const __bf16* Bt, __bf16* C,
                     int m, int n, int k) {
    dim3 grid(n / BN, m / BM);
    gemm_bf16_kernel<__bf16><<<grid, 256, 0, stream>>>(A, Bt, C, m, n, k);
  };

  gemm_bf(xb,  Wqt,  qb,  M, D, D);   // q      = x  @ Wq
  gemm_bf(xb,  Wkvt, kvb, M, L, D);   // kv_lat = x  @ Wkv
  gemm_bf(kvb, Wkt,  kb,  M, D, L);   // k      = kv @ Wk
  gemm_bf(kvb, Wvt,  vb,  M, D, L);   // v      = kv @ Wv

  dim3 agrid(S / 16, H, B);
  attn_window_kernel<<<agrid, 128, 0, stream>>>(qb, kb, vb, aob, S, H);

  dim3 ogrid(D / BN, M / BM);
  gemm_bf16_kernel<float><<<ogrid, 256, 0, stream>>>(aob, Wot, (float*)d_out,
                                                     M, D, D);
}
